// LoRALinear_11673721110784
// MI455X (gfx1250) — matrix operand order repaired and checked by reading, not compiled, not run on hardware
//
#include <hip/hip_runtime.h>

// ---------------------------------------------------------------------------
// LoRA linear on gfx1250 (MI455X), compile-only tuned.
//   out[M,N] = x[M,K] @ W_eff[N,K]^T + bias,  W_eff = W + 2*(lora_B @ lora_A)
// Main GEMM: v_wmma_f32_16x16x32_bf16 (wave32). Operand tiles are staged
// into LDS by the Tensor Data Mover (tensor_load_to_lds, TENSORcnt,
// double-buffered) when the builtin is available; otherwise fragments are
// loaded directly from row-major global memory.
// ---------------------------------------------------------------------------

typedef __attribute__((ext_vector_type(16))) __bf16       v16bf;
typedef __attribute__((ext_vector_type(8)))  float        v8f;
typedef __attribute__((ext_vector_type(4)))  unsigned int u32x4;
typedef __attribute__((ext_vector_type(2)))  unsigned int u32x2;
typedef __attribute__((ext_vector_type(4)))  int          i32x4;
typedef __attribute__((ext_vector_type(8)))  int          i32x8;

union FragBF {
    v16bf v;
    u32x4 q[2];
};

// Per-lane WMMA fragment = two contiguous 16B runs (A and B: second at +16
// halves). Works for both global and LDS source pointers.
__device__ __forceinline__ v16bf load_frag(const __bf16* __restrict__ p,
                                           int second_off) {
    FragBF f;
    f.q[0] = *reinterpret_cast<const u32x4*>(p);
    f.q[1] = *reinterpret_cast<const u32x4*>(p + second_off);
    return f.v;
}

// --------------------------- kernel 1: x -> bf16 ---------------------------
__global__ __launch_bounds__(256) void cvt_f32_to_bf16(
    const float* __restrict__ in, __bf16* __restrict__ out, long n) {
    long i = ((long)blockIdx.x * 256 + threadIdx.x) * 8;
    if (i + 8 > n) return;
    float4 a = *reinterpret_cast<const float4*>(in + i);
    float4 b = *reinterpret_cast<const float4*>(in + i + 4);
    union { __bf16 h[8]; u32x4 q; } u;
    u.h[0] = (__bf16)a.x; u.h[1] = (__bf16)a.y;
    u.h[2] = (__bf16)a.z; u.h[3] = (__bf16)a.w;
    u.h[4] = (__bf16)b.x; u.h[5] = (__bf16)b.y;
    u.h[6] = (__bf16)b.z; u.h[7] = (__bf16)b.w;
    *reinterpret_cast<u32x4*>(out + i) = u.q;
}

// ----------------- kernel 2: W_eff = W + 2*(B@A), cast bf16 ----------------
__global__ __launch_bounds__(256) void build_weff(
    const float* __restrict__ w,      // [N,K]
    const float* __restrict__ la,     // [R,K]
    const float* __restrict__ lb,     // [N,R]
    __bf16* __restrict__ out,         // [N,K] bf16
    int K, int R) {
    const int o  = blockIdx.y;
    const long d = ((long)blockIdx.x * 256 + threadIdx.x) * 4;
    if (d + 4 > K) return;
    float4 acc = *reinterpret_cast<const float4*>(w + (long)o * K + d);
    for (int r = 0; r < R; ++r) {
        const float s = 2.0f * lb[o * R + r];           // scalar (SGPR) load
        const float4 av = *reinterpret_cast<const float4*>(la + (long)r * K + d);
        acc.x += s * av.x; acc.y += s * av.y;
        acc.z += s * av.z; acc.w += s * av.w;
    }
    union { __bf16 h[4]; u32x2 q; } u;
    u.h[0] = (__bf16)acc.x; u.h[1] = (__bf16)acc.y;
    u.h[2] = (__bf16)acc.z; u.h[3] = (__bf16)acc.w;
    *reinterpret_cast<u32x2*>(out + (long)o * K + d) = u.q;
}

// ------------------------- Tensor Data Mover setup -------------------------
#if __has_builtin(__builtin_amdgcn_tensor_load_to_lds)
#define LORA_HAVE_TDM 1
#else
#define LORA_HAVE_TDM 0
#endif

// LDS tile: 128 rows x 32 bf16, padded by TDM to an 80B row pitch
// (pad_interval = 16 DWORDs = 64B = one tile row, pad_amount = 4 DWORDs).
// Pitch 80B => per-lane ds_load_b128 start banks are 20*c mod 64: all
// distinct, stride 4 -> conflict-free within each half-wave.
#define LDSROW 40            // elements per LDS row (32 data + 4 DW pad)
#define TILE_ELEMS (128 * LDSROW)

#if LORA_HAVE_TDM
// DMA one 128-row x 32-col bf16 tile (row stride = row_stride_elems) from
// global to LDS at byte offset lds_off. D# per CDNA5 ISA 8.3/8.4.
__device__ __forceinline__ void tdm_load_tile_128x32(
    const __bf16* __restrict__ g, unsigned lds_off, int row_stride_elems) {
    unsigned long long ga = (unsigned long long)g;
    u32x4 g0;
    g0[0] = 1u;                                          // count=1, user D#
    g0[1] = lds_off;                                     // lds_addr
    g0[2] = (unsigned)ga;                                // global_addr[31:0]
    g0[3] = (unsigned)((ga >> 32) & 0x1FFFFFFull)        // global_addr[56:32]
            | (2u << 30);                                // type = 2 (image)

    i32x8 g1;
    g1[0] = (1 << 16)            // data_size = 1 -> 2 bytes (bf16)
          | (1 << 20)            // pad_enable
          | (3 << 22)            // pad_interval: 16 DWORDs (64B)
          | (3 << 25);           // pad_amount:   4 DWORDs (16B)
    g1[1] = (32 & 0xFFFF) << 16;                  // tensor_dim0 = 32 (lo16)
    g1[2] = ((32 >> 16) & 0xFFFF)                 // tensor_dim0 (hi16) = 0
          | ((128 & 0xFFFF) << 16);               // tensor_dim1 = 128 (lo16)
    g1[3] = ((128 >> 16) & 0xFFFF)                // tensor_dim1 (hi16) = 0
          | (32 << 16);                           // tile_dim0 = 32
    g1[4] = 128;                                  // tile_dim1 = 128, dim2 = 0
    g1[5] = row_stride_elems;                     // tensor_dim0_stride lo32
    g1[6] = 0;                                    // stride hi16, dim1_stride lo
    g1[7] = 0;
    i32x4 z4 = {0, 0, 0, 0};
#if defined(__clang_major__) && (__clang_major__ >= 23)
    i32x8 z8 = {0, 0, 0, 0, 0, 0, 0, 0};
    __builtin_amdgcn_tensor_load_to_lds(g0, g1, z4, z4, z8, 0);
#else
    __builtin_amdgcn_tensor_load_to_lds(g0, g1, z4, z4, 0);
#endif
}
#endif // LORA_HAVE_TDM

// --------------------- kernel 3: WMMA bf16 GEMM + bias ---------------------
// Block = 128 threads = 4 waves (2x2); wave tile 64x64 = 4x4 WMMA subtiles.
// Block tile 128x128; K-loop in steps of 32, LDS double-buffered via TDM.
__global__ __launch_bounds__(128) void lora_gemm_wmma(
    const __bf16* __restrict__ X,     // [M,K] bf16 row-major
    const __bf16* __restrict__ W,     // [N,K] bf16 row-major (acts as B^T)
    const float*  __restrict__ bias,  // [N]
    float*        __restrict__ out,   // [M,N] f32
    int M, int N, int K) {
    const int lane = threadIdx.x & 31;
    const int wave = threadIdx.x >> 5;
    const int mB = blockIdx.y * 128;
    const int nB = blockIdx.x * 128;
    const int wmRow = (wave >> 1) * 64;
    const int wnCol = (wave & 1) * 64;

    const int row16 = lane & 15;
    const int kaoff = (lane & 16) ? 8 : 0;    // A: lanes 16-31 start at K+8
    const int kboff = (lane & 16) ? 8 : 0;    // B: lanes 16-31 start at K+8

    v8f acc[4][4] = {};

#if LORA_HAVE_TDM
    __shared__ __bf16 sA[2][TILE_ELEMS];
    __shared__ __bf16 sB[2][TILE_ELEMS];

    // Per-lane element offsets inside an LDS tile.
    int aoff[4], boff[4];
#pragma unroll
    for (int i = 0; i < 4; ++i)
        aoff[i] = (wmRow + i * 16 + row16) * LDSROW + kaoff;
#pragma unroll
    for (int j = 0; j < 4; ++j)
        boff[j] = (wnCol + j * 16 + row16) * LDSROW + kboff;

    const unsigned ldsA0 = (unsigned)(unsigned long long)&sA[0][0];
    const unsigned ldsA1 = (unsigned)(unsigned long long)&sA[1][0];
    const unsigned ldsB0 = (unsigned)(unsigned long long)&sB[0][0];
    const unsigned ldsB1 = (unsigned)(unsigned long long)&sB[1][0];

    const int nsteps = K / 32;
    if (wave == 0) {
        tdm_load_tile_128x32(X + (long)mB * K, ldsA0, K);
        tdm_load_tile_128x32(W + (long)nB * K, ldsB0, K);
    }

    for (int s = 0; s < nsteps; ++s) {
        const int cur = s & 1;
        if (wave == 0) __builtin_amdgcn_s_wait_tensorcnt(0);
        __syncthreads();                       // buf[cur] ready for everyone
        if ((wave == 0) && (s + 1 < nsteps)) { // prefetch buf[1-cur]
            const int k = (s + 1) * 32;
            tdm_load_tile_128x32(X + (long)mB * K + k, cur ? ldsA0 : ldsA1, K);
            tdm_load_tile_128x32(W + (long)nB * K + k, cur ? ldsB0 : ldsB1, K);
        }
        const __bf16* __restrict__ pA = &sA[cur][0];
        const __bf16* __restrict__ pB = &sB[cur][0];
        v16bf a[4], b[4];
#pragma unroll
        for (int i = 0; i < 4; ++i) a[i] = load_frag(pA + aoff[i], 16);
#pragma unroll
        for (int j = 0; j < 4; ++j) b[j] = load_frag(pB + boff[j], 16);
#pragma unroll
        for (int i = 0; i < 4; ++i)
#pragma unroll
            for (int j = 0; j < 4; ++j)
                acc[i][j] = __builtin_amdgcn_wmma_f32_16x16x32_bf16(
                    false, a[i], false, b[j], (short)0, acc[i][j],
                    false, false);
    }
#else
    // Fallback: fragments straight from global.
    const __bf16* pa[4];
    const __bf16* pb[4];
#pragma unroll
    for (int i = 0; i < 4; ++i)
        pa[i] = X + (long)(mB + wmRow + i * 16 + row16) * K + kaoff;
#pragma unroll
    for (int j = 0; j < 4; ++j)
        pb[j] = W + (long)(nB + wnCol + j * 16 + row16) * K + kboff;

#pragma unroll 2
    for (int k = 0; k < K; k += 32) {
        v16bf a[4], b[4];
#pragma unroll
        for (int i = 0; i < 4; ++i) a[i] = load_frag(pa[i] + k, 16);
#pragma unroll
        for (int j = 0; j < 4; ++j) b[j] = load_frag(pb[j] + k, 16);
#pragma unroll
        for (int i = 0; i < 4; ++i)
#pragma unroll
            for (int j = 0; j < 4; ++j)
                acc[i][j] = __builtin_amdgcn_wmma_f32_16x16x32_bf16(
                    false, a[i], false, b[j], (short)0, acc[i][j],
                    false, false);
    }
#endif

    // Epilogue: C/D layout -> lanes 0-15: N=lane, M=vgpr; lanes 16-31: M=8+vgpr.
    const int nl   = lane & 15;
    const int moff = (lane & 16) ? 8 : 0;
#pragma unroll
    for (int j = 0; j < 4; ++j) {
        const int col = nB + wnCol + j * 16 + nl;
        const float bv = bias[col];
#pragma unroll
        for (int i = 0; i < 4; ++i) {
#pragma unroll
            for (int e = 0; e < 8; ++e) {
                out[(long)(mB + wmRow + i * 16 + moff + e) * N + col] =
                    acc[i][j][e] + bv;
            }
        }
    }
}

// ---------------------------------------------------------------------------
extern "C" void kernel_launch(void* const* d_in, const int* in_sizes, int n_in,
                              void* d_out, int out_size, void* d_ws,
                              size_t ws_size, hipStream_t stream) {
    const float* x    = (const float*)d_in[0];  // [B,S,K]
    const float* w    = (const float*)d_in[1];  // [N,K]
    const float* bias = (const float*)d_in[2];  // [N]
    const float* la   = (const float*)d_in[3];  // [R,K]
    const float* lb   = (const float*)d_in[4];  // [N,R]

    const int  N  = in_sizes[2];                 // 4096
    const int  K  = in_sizes[1] / N;             // 4096
    const int  R  = in_sizes[3] / K;             // 16
    const long xN = (long)in_sizes[0];           // M*K
    const long M  = xN / K;                      // 8192

    // Workspace: [0, xN*2) = x in bf16; then N*K*2 bytes = W_eff in bf16.
    __bf16* xb = (__bf16*)d_ws;
    __bf16* wb = xb + xN;

    // 1) x -> bf16
    cvt_f32_to_bf16<<<dim3((unsigned)((xN / 8 + 255) / 256)), 256, 0, stream>>>(
        x, xb, xN);

    // 2) W_eff = W + 2*(lora_B @ lora_A), stored bf16
    build_weff<<<dim3((unsigned)((K / 4 + 255) / 256), (unsigned)N), 256, 0,
                 stream>>>(w, la, lb, wb, K, R);

    // 3) out = x @ W_eff^T + bias   (M,N,K multiples of 128/128/32)
    lora_gemm_wmma<<<dim3((unsigned)(N / 128), (unsigned)(M / 128)), 128, 0,
                     stream>>>(xb, wb, bias, (float*)d_out, (int)M, N, K);
}
